// Autoregressive2dJoints_54417235640682
// MI455X (gfx1250) — compile-verified, matrix operand order checked by reading, not run on hardware
//
#include <hip/hip_runtime.h>

// ---------------------------------------------------------------------------
// Autoregressive LSTM (encode -> LSTM -> decode, 10 seed + 50 AR steps)
// B=16384 T=60 D=34 E=128 H=64.  Batch-parallel persistent kernel:
// each wave32 owns 16 batch rows, carries (h,c) across all steps, and does
// every matmul as v_wmma_f32_16x16x32_bf16 (M=16 batch rows, f32 accum).
// ---------------------------------------------------------------------------

typedef __bf16 v16bf __attribute__((ext_vector_type(16)));
typedef float  v8f   __attribute__((ext_vector_type(8)));

#define BB     16384
#define TT     60
#define DD     34
#define EE     128
#define HH     64
#define GG     256     // 4*H
#define NSEED  10
#define NPRED  50

// --- weight-fragment region layout (identical in d_ws and LDS), bytes ------
#define WIH_OFF   0        // W_ih^T  (E x 4H):  4 ktiles x 16 ntiles x 1KB = 64KB
#define WHH_OFF   65536    // W_hh^T  (H x 4H):  2 x 16 x 1KB = 32KB
#define WENC_OFF  98304    // W_enc^T (64pad x E): 2 x 8 x 1KB = 16KB
#define WDEC_OFF  114688   // W_dec^T (H x 48pad): 2 x 3 x 1KB = 6KB
#define BG_OFF    120832   // b_ih + b_hh (256 f32)
#define BE_OFF    121856   // b_enc (128 f32)
#define BD_OFF    122368   // b_dec padded to 48 f32
#define WBYTES    122560   // multiple of 16
#define PW_BASE   122880   // per-wave scratch base (1KB aligned)
#define PW_STRIDE 8192     // inpA(4KB) + hA(2KB) + decA(2KB)
#define LDS_BYTES (PW_BASE + 4 * PW_STRIDE)   // 155648 B

// --- math helpers ----------------------------------------------------------
__device__ __forceinline__ float sigm(float v)  { return 1.0f / (1.0f + __expf(-v)); }
__device__ __forceinline__ float tanh_(float v) { return 2.0f * sigm(2.0f * v) - 1.0f; }

__device__ __forceinline__ v8f wmma_bf16(v16bf a, v16bf b, v8f c) {
  // (neg_a, A, neg_b, B, c_mod, C, reuse_a, reuse_b)
  return __builtin_amdgcn_wmma_f32_16x16x32_bf16(false, a, false, b, (short)0, c,
                                                 false, false);
}

// Byte offset of element (m, k) inside an A-fragment region laid out as
// ktile-major, 32 lanes x 32B per fragment, following the documented
// 16-bit 16x32 A-matrix VGPR mapping (ISA 7.12.2).
__device__ __forceinline__ int afrag_addr(int m, int k) {
  int kt  = k >> 5;
  int kl  = k & 31;
  int hi  = (kl >> 4) & 1;       // K>=16 half -> VGPR 4..7
  int rem = kl & 15;
  int g   = rem >> 3;            // lane group 0..1
  int r8  = rem & 7;
  int v   = hi * 4 + (r8 >> 1);  // VGPR index 0..7
  int hh  = r8 & 1;              // packed half
  int lane = g * 16 + m;
  return (kt << 10) + lane * 32 + v * 4 + hh * 2;
}

// ---------------------------------------------------------------------------
// prep kernel: swizzle f32 weights into bf16 WMMA B-fragment order in d_ws.
// B fragment mapping mirrors the A mapping with (row->K, lane%16 -> N col).
// ---------------------------------------------------------------------------
__global__ void prep_weights(const float* __restrict__ W_enc,
                             const float* __restrict__ W_ih,
                             const float* __restrict__ W_hh,
                             const float* __restrict__ W_dec,
                             const float* __restrict__ b_ih,
                             const float* __restrict__ b_hh,
                             const float* __restrict__ b_enc,
                             const float* __restrict__ b_dec,
                             unsigned char* __restrict__ ws) {
  int e = blockIdx.x * blockDim.x + threadIdx.x;

  auto dekl = [](int within, int& kl, int& nl) {
    int lane = within >> 4;
    int v16  = within & 15;
    int v  = v16 >> 1;
    int hh = v16 & 1;
    int g  = lane >> 4;
    kl = ((v >> 2) << 4) + g * 8 + ((v & 3) << 1) + hh;
    nl = lane & 15;
  };

  if (e < 32768) {                                    // W_ih^T  frags (kt*16+nt)
    int frag = e >> 9, within = e & 511;
    int kt = frag >> 4, nt = frag & 15;
    int kl, nl; dekl(within, kl, nl);
    int k = kt * 32 + kl;                             // E index
    int n = nt * 16 + nl;                             // gate index
    ((__bf16*)(ws + WIH_OFF))[e] = (__bf16)W_ih[n * EE + k];
  } else if (e < 49152) {                             // W_hh^T
    int le = e - 32768;
    int frag = le >> 9, within = le & 511;
    int kt = frag >> 4, nt = frag & 15;
    int kl, nl; dekl(within, kl, nl);
    int k = kt * 32 + kl;                             // H index
    int n = nt * 16 + nl;                             // gate index
    ((__bf16*)(ws + WHH_OFF))[le] = (__bf16)W_hh[n * HH + k];
  } else if (e < 57344) {                             // W_enc^T, K padded 34->64
    int le = e - 49152;
    int frag = le >> 9, within = le & 511;
    int kt = frag >> 3, nt = frag & 7;
    int kl, nl; dekl(within, kl, nl);
    int k = kt * 32 + kl;                             // D index (padded)
    int n = nt * 16 + nl;                             // E index
    float v = (k < DD) ? W_enc[n * DD + k] : 0.0f;
    ((__bf16*)(ws + WENC_OFF))[le] = (__bf16)v;
  } else if (e < 60416) {                             // W_dec^T, N padded 34->48
    int le = e - 57344;
    int frag = le >> 9, within = le & 511;
    int kt = frag / 3, nt = frag % 3;
    int kl, nl; dekl(within, kl, nl);
    int k = kt * 32 + kl;                             // H index
    int n = nt * 16 + nl;                             // D index (padded)
    float v = (n < DD) ? W_dec[n * HH + k] : 0.0f;
    ((__bf16*)(ws + WDEC_OFF))[le] = (__bf16)v;
  } else {                                            // biases
    int le = e - 60416;
    if (le < 256) {
      ((float*)(ws + BG_OFF))[le] = b_ih[le] + b_hh[le];
    } else if (le < 384) {
      ((float*)(ws + BE_OFF))[le - 256] = b_enc[le - 256];
    } else if (le < 432) {
      int i = le - 384;
      ((float*)(ws + BD_OFF))[i] = (i < DD) ? b_dec[i] : 0.0f;
    }
  }
}

// ---------------------------------------------------------------------------
// main persistent kernel: 128 threads (4 waves), 64 batch rows per block.
// ---------------------------------------------------------------------------
__global__ __launch_bounds__(128) void ar_lstm(const float* __restrict__ x,
                                               const unsigned char* __restrict__ wsw,
                                               float* __restrict__ out) {
  extern __shared__ char smem[];
  const int  lane = threadIdx.x & 31;
  const int  wave = threadIdx.x >> 5;
  const int  l15  = lane & 15;
  const int  mrow = (lane >> 4) << 3;                 // +8 rows for upper group
  const long rowbase = (long)blockIdx.x * 64 + wave * 16;

  // cooperative linear copy of all weight fragments + biases into LDS
  {
    const uint4* src = (const uint4*)wsw;
    uint4* dst = (uint4*)smem;
    for (int i = threadIdx.x; i < WBYTES / 16; i += blockDim.x) dst[i] = src[i];
  }
  __syncthreads();

  char* pw   = smem + PW_BASE + wave * PW_STRIDE;
  char* inpA = pw;                // 4 frags: rnn_in (16 x 128 bf16), A layout
  char* hA   = pw + 4096;         // 2 frags: h      (16 x 64  bf16), A layout
  char* decA = pw + 6144;         // 2 frags: dec/x  (16 x 64  bf16, padded)

  // zero h and decA (padding columns >= 34 are written exactly once, here)
  {
    uint4 z = {0u, 0u, 0u, 0u};
#pragma unroll
    for (int f = 0; f < 2; ++f) {
      *(uint4*)(hA   + f * 1024 + lane * 32)      = z;
      *(uint4*)(hA   + f * 1024 + lane * 32 + 16) = z;
      *(uint4*)(decA + f * 1024 + lane * 32)      = z;
      *(uint4*)(decA + f * 1024 + lane * 32 + 16) = z;
    }
  }

  const float* bgate = (const float*)(smem + BG_OFF);
  const float* benc  = (const float*)(smem + BE_OFF);
  const float* bdec  = (const float*)(smem + BD_OFF);

  v8f cacc[4];
#pragma unroll
  for (int j = 0; j < 4; ++j)
#pragma unroll
    for (int e = 0; e < 8; ++e) cacc[j][e] = 0.0f;

  auto splat = [&](const float* p, int col0) -> v8f {
    float bv = p[col0 + l15];
    v8f r;
#pragma unroll
    for (int e = 0; e < 8; ++e) r[e] = bv;
    return r;
  };
  auto ldA = [&](const char* base, int kt) -> v16bf {
    return *(const v16bf*)(base + (kt << 10) + lane * 32);
  };
  auto ldB = [&](int off, int nt16, int kt, int nt) -> v16bf {
    return *(const v16bf*)(smem + off + ((kt * nt16 + nt) << 10) + lane * 32);
  };

  // one LSTM step: gates = rnn_in @ W_ih^T + h @ W_hh^T + (b_ih+b_hh)
  auto lstm_step = [&]() {
    v16bf ai[4], ah[2];
#pragma unroll
    for (int kt = 0; kt < 4; ++kt) ai[kt] = ldA(inpA, kt);
#pragma unroll
    for (int kt = 0; kt < 2; ++kt) ah[kt] = ldA(hA, kt);
#pragma unroll
    for (int jt = 0; jt < 4; ++jt) {          // H column tiles of 16
      v8f gi = splat(bgate, jt * 16);
      v8f gf = splat(bgate, 64 + jt * 16);
      v8f gg = splat(bgate, 128 + jt * 16);
      v8f go = splat(bgate, 192 + jt * 16);
#pragma unroll
      for (int kt = 0; kt < 4; ++kt) {        // K over E=128
        gi = wmma_bf16(ai[kt], ldB(WIH_OFF, 16, kt, jt),      gi);
        gf = wmma_bf16(ai[kt], ldB(WIH_OFF, 16, kt, 4 + jt),  gf);
        gg = wmma_bf16(ai[kt], ldB(WIH_OFF, 16, kt, 8 + jt),  gg);
        go = wmma_bf16(ai[kt], ldB(WIH_OFF, 16, kt, 12 + jt), go);
      }
#pragma unroll
      for (int kt = 0; kt < 2; ++kt) {        // K over H=64
        gi = wmma_bf16(ah[kt], ldB(WHH_OFF, 16, kt, jt),      gi);
        gf = wmma_bf16(ah[kt], ldB(WHH_OFF, 16, kt, 4 + jt),  gf);
        gg = wmma_bf16(ah[kt], ldB(WHH_OFF, 16, kt, 8 + jt),  gg);
        go = wmma_bf16(ah[kt], ldB(WHH_OFF, 16, kt, 12 + jt), go);
      }
#pragma unroll
      for (int e = 0; e < 8; ++e) {           // elementwise cell update
        float iv = sigm(gi[e]);
        float fv = sigm(gf[e]);
        float gv = tanh_(gg[e]);
        float ov = sigm(go[e]);
        float cn = fv * cacc[jt][e] + iv * gv;
        cacc[jt][e] = cn;
        float hn = ov * tanh_(cn);
        *(__bf16*)(hA + afrag_addr(e + mrow, jt * 16 + l15)) = (__bf16)hn;
      }
    }
  };

  // rnn_in = relu(srcA @ W_enc^T + b_enc) -> inpA fragments
  auto encode = [&](const char* srcA) {
    v16bf a0 = ldA(srcA, 0), a1 = ldA(srcA, 1);
#pragma unroll
    for (int nt = 0; nt < 8; ++nt) {
      v8f acc = splat(benc, nt * 16);
      acc = wmma_bf16(a0, ldB(WENC_OFF, 8, 0, nt), acc);
      acc = wmma_bf16(a1, ldB(WENC_OFF, 8, 1, nt), acc);
#pragma unroll
      for (int e = 0; e < 8; ++e) {
        float v = acc[e] > 0.0f ? acc[e] : 0.0f;
        *(__bf16*)(inpA + afrag_addr(e + mrow, nt * 16 + l15)) = (__bf16)v;
      }
    }
  };

  // dec = h @ W_dec^T + b_dec  (3 C/D tiles, cols >= 34 are pad)
  auto decode = [&](v8f dec[3]) {
    v16bf a0 = ldA(hA, 0), a1 = ldA(hA, 1);
#pragma unroll
    for (int nt = 0; nt < 3; ++nt) {
      v8f acc = splat(bdec, nt * 16);
      acc = wmma_bf16(a0, ldB(WDEC_OFF, 3, 0, nt), acc);
      acc = wmma_bf16(a1, ldB(WDEC_OFF, 3, 1, nt), acc);
      dec[nt] = acc;
    }
  };

  auto dec_to_A = [&](const v8f dec[3]) {     // C/D layout -> decA fragments
#pragma unroll
    for (int nt = 0; nt < 3; ++nt) {
      int k = nt * 16 + l15;
      if (k < DD) {
#pragma unroll
        for (int e = 0; e < 8; ++e)
          *(__bf16*)(decA + afrag_addr(e + mrow, k)) = (__bf16)dec[nt][e];
      }
    }
  };

  auto load_xA = [&](int t) {                 // x[:, t, :] -> decA fragments
    for (int idx = lane; idx < 16 * DD; idx += 32) {
      int m = idx / DD, d = idx % DD;
      float v = x[((rowbase + m) * TT + t) * (long)DD + d];
      *(__bf16*)(decA + afrag_addr(m, d)) = (__bf16)v;
    }
  };

  // ---------------- seed phase --------------------------------------------
#pragma unroll 1
  for (int t = 0; t < NSEED; ++t) {
    if (t + 1 < NSEED)
      __builtin_prefetch(x + ((rowbase + lane) * TT + t + 1) * (long)DD, 0, 1);
    load_xA(t);
    encode(decA);
    lstm_step();
  }

  // dec_out = decode(h); rnn_in = encode(dec_out); prev = x[:, NSEED-1, :]
  v8f dec[3];
  decode(dec);
  dec_to_A(dec);
  encode(decA);

  v8f prev[3];
#pragma unroll
  for (int nt = 0; nt < 3; ++nt) {
    int d = nt * 16 + l15;
#pragma unroll
    for (int e = 0; e < 8; ++e)
      prev[nt][e] = (d < DD)
          ? x[((rowbase + e + mrow) * TT + (NSEED - 1)) * (long)DD + d]
          : 0.0f;
  }

  // ---------------- autoregressive phase ----------------------------------
#pragma unroll 1
  for (int s = 0; s < NPRED; ++s) {
    lstm_step();
    decode(dec);
#pragma unroll
    for (int nt = 0; nt < 3; ++nt) {
      int d = nt * 16 + l15;
#pragma unroll
      for (int e = 0; e < 8; ++e) {
        float o = dec[nt][e] + prev[nt][e];
        prev[nt][e] = o;
        if (d < DD)
          out[((rowbase + e + mrow) * (long)NPRED + s) * DD + d] = o;
      }
    }
    dec_to_A(dec);
    encode(decA);
  }
}

// ---------------------------------------------------------------------------
extern "C" void kernel_launch(void* const* d_in, const int* in_sizes, int n_in,
                              void* d_out, int out_size, void* d_ws, size_t ws_size,
                              hipStream_t stream) {
  const float* x     = (const float*)d_in[0];
  const float* W_enc = (const float*)d_in[1];
  const float* b_enc = (const float*)d_in[2];
  const float* W_ih  = (const float*)d_in[3];
  const float* W_hh  = (const float*)d_in[4];
  const float* b_ih  = (const float*)d_in[5];
  const float* b_hh  = (const float*)d_in[6];
  const float* W_dec = (const float*)d_in[7];
  const float* b_dec = (const float*)d_in[8];
  float* out = (float*)d_out;
  unsigned char* ws = (unsigned char*)d_ws;

  // 60416 bf16 fragment elements + 432 bias floats
  prep_weights<<<238, 256, 0, stream>>>(W_enc, W_ih, W_hh, W_dec,
                                        b_ih, b_hh, b_enc, b_dec, ws);
  ar_lstm<<<BB / 64, 128, LDS_BYTES, stream>>>(x, ws, out);
}
